// AVFusion_50130858278971
// MI455X (gfx1250) — compile-verified, matrix operand order checked by reading, not run on hardware
//
#include <hip/hip_runtime.h>
#include <hip/hip_bf16.h>

typedef __attribute__((ext_vector_type(16))) __bf16 v16bf;
typedef __attribute__((ext_vector_type(8)))  float  v8f;

#define BSZ  16
#define NSEG 64
#define NSEN 32
#define DM   1024
#define DFF  2048

// ---------------- elementwise / prep kernels ----------------

// W: K x N (in,out) row-major  ->  Wt: N x K row-major (= W^T), fp32 -> bf16
__global__ __launch_bounds__(256)
void k_transpose_bf16(const float* __restrict__ W, __bf16* __restrict__ Wt,
                      int K, int N) {
  long i = (long)blockIdx.x * 256 + threadIdx.x;
  long total = (long)K * N;
  if (i >= total) return;
  int n = (int)(i / K);
  int k = (int)(i % K);
  Wt[i] = (__bf16)W[(long)k * N + n];
}

// av[b,seg,0,:] = A[b,seg,:], av[b,seg,1,:] = V[b,seg,:]  (bf16)
__global__ __launch_bounds__(256)
void k_build_av(const float* __restrict__ A, const float* __restrict__ V,
                __bf16* __restrict__ av, long rows, int d) {
  long i = (long)blockIdx.x * 256 + threadIdx.x;
  long total = rows * (long)d;
  if (i >= total) return;
  long r = i / d; int c = (int)(i % d);
  av[(2*r)   * (long)d + c] = (__bf16)A[i];
  av[(2*r+1) * (long)d + c] = (__bf16)V[i];
}

// per-row layernorm, fp32 in -> bf16 out. One block (256 thr = 8 waves) per row.
__global__ __launch_bounds__(256)
void k_layernorm_bf16(const float* __restrict__ S, const float* __restrict__ gamma,
                      const float* __restrict__ beta, __bf16* __restrict__ out, int D) {
  __shared__ float red0[8], red1[8];
  const float* x = S + (long)blockIdx.x * D;
  float s = 0.f, s2 = 0.f;
  for (int i = threadIdx.x; i < D; i += 256) { float v = x[i]; s += v; s2 += v * v; }
  for (int o = 16; o > 0; o >>= 1) { s += __shfl_xor(s, o, 32); s2 += __shfl_xor(s2, o, 32); }
  if ((threadIdx.x & 31) == 0) { red0[threadIdx.x >> 5] = s; red1[threadIdx.x >> 5] = s2; }
  __syncthreads();
  if (threadIdx.x == 0) {
    float a = 0.f, b = 0.f;
    for (int w = 0; w < 8; ++w) { a += red0[w]; b += red1[w]; }
    red0[0] = a; red1[0] = b;
  }
  __syncthreads();
  float mu   = red0[0] / D;
  float var  = red1[0] / D - mu * mu;
  float rstd = rsqrtf(var + 1e-5f);
  __bf16* o = out + (long)blockIdx.x * D;
  for (int i = threadIdx.x; i < D; i += 256)
    o[i] = (__bf16)((x[i] - mu) * rstd * gamma[i] + beta[i]);
}

// ---------------- WMMA bf16 GEMM:  C(MxN) = A(MxK) * Bt(NxK)^T + bias ----------------
// 128 threads (4 waves). Block tile 64x64, K-step 32, double-buffered LDS filled by
// GLOBAL_LOAD_ASYNC_TO_LDS_B128 (ASYNCcnt). Wave w -> rows [w*16, w*16+16), 4 accs
// across the 64 output columns (A fragment reused 4x).

template<int RELU, typename OutT>
__global__ __launch_bounds__(128)
void k_gemm_bf16nt(const __bf16* __restrict__ A, const __bf16* __restrict__ Bt,
                   const float* __restrict__ bias, OutT* __restrict__ C,
                   int M, int N, int K) {
  constexpr int BM = 64, BN = 64, BK = 32, LDT = 48; // 96B row stride: 16B-aligned
  __shared__ __align__(16) __bf16 As[2][BM][LDT];
  __shared__ __align__(16) __bf16 Bs[2][BN][LDT];

  const int tid   = threadIdx.x;
  const int wave  = tid >> 5;
  const int lane  = tid & 31;
  const int lrow  = lane & 15;
  const int khalf = (lane >> 4) << 3;            // 16-bit A layout: lanes 0-15 K=0..7, 16-31 K=8..15
  const long m_blk = (long)blockIdx.y * BM;
  const long n_blk = (long)blockIdx.x * BN;

  // this thread's two 16B staging chunks (same (r,c) for A and B tiles)
  const int ch0 = tid, ch1 = tid + 128;
  const int r0 = ch0 >> 2, c0 = (ch0 & 3) << 3;
  const int r1 = ch1 >> 2, c1 = (ch1 & 3) << 3;
  const __bf16* gA0 = &A [(m_blk + r0) * (long)K + c0];
  const __bf16* gA1 = &A [(m_blk + r1) * (long)K + c1];
  const __bf16* gB0 = &Bt[(n_blk + r0) * (long)K + c0];
  const __bf16* gB1 = &Bt[(n_blk + r1) * (long)K + c1];

  v8f acc[4] = {};

  // async DMA of one 64x32 K-tile pair into LDS buffer `buf`
  auto issue_tile = [&](int k0, int buf) {
    unsigned lA0 = (unsigned)(size_t)&As[buf][r0][c0];
    unsigned lA1 = (unsigned)(size_t)&As[buf][r1][c1];
    unsigned lB0 = (unsigned)(size_t)&Bs[buf][r0][c0];
    unsigned lB1 = (unsigned)(size_t)&Bs[buf][r1][c1];
    asm volatile("global_load_async_to_lds_b128 %0, %1, off" :: "v"(lA0), "v"(gA0 + k0) : "memory");
    asm volatile("global_load_async_to_lds_b128 %0, %1, off" :: "v"(lA1), "v"(gA1 + k0) : "memory");
    asm volatile("global_load_async_to_lds_b128 %0, %1, off" :: "v"(lB0), "v"(gB0 + k0) : "memory");
    asm volatile("global_load_async_to_lds_b128 %0, %1, off" :: "v"(lB1), "v"(gB1 + k0) : "memory");
  };

  issue_tile(0, 0);
  asm volatile("s_wait_asynccnt 0x0" ::: "memory");
  __syncthreads();

  int cur = 0;
  for (int k0 = 0; k0 < K; k0 += BK) {
    const int nxt = cur ^ 1;
    if (k0 + BK < K) {
      issue_tile(k0 + BK, nxt);                   // overlap DMA with WMMA below
      __builtin_prefetch(gA0 + k0 + 2 * BK, 0, 1);  // speculative: dropped if unmapped
      __builtin_prefetch(gB0 + k0 + 2 * BK, 0, 1);
    }

    v16bf a;
    #pragma unroll
    for (int e = 0; e < 8; ++e) {
      a[e]     = As[cur][wave * 16 + lrow][khalf + e];
      a[e + 8] = As[cur][wave * 16 + lrow][16 + khalf + e];
    }
    #pragma unroll
    for (int j = 0; j < 4; ++j) {
      v16bf b;
      #pragma unroll
      for (int e = 0; e < 8; ++e) {
        b[e]     = Bs[cur][j * 16 + lrow][khalf + e];
        b[e + 8] = Bs[cur][j * 16 + lrow][16 + khalf + e];
      }
      acc[j] = __builtin_amdgcn_wmma_f32_16x16x32_bf16(
                   false, a, false, b, (short)0, acc[j], false, false);
    }

    asm volatile("s_wait_asynccnt 0x0" ::: "memory");
    __syncthreads();                              // buffer `nxt` ready; `cur` free to overwrite
    cur = nxt;
  }

  // C layout: vgpr r, lane l -> row = r + 8*(l>=16), col = l&15
  const long mr  = m_blk + wave * 16 + ((lane >> 4) << 3);
  const int  nc0 = (int)n_blk + (lane & 15);
  #pragma unroll
  for (int j = 0; j < 4; ++j) {
    int   nc = nc0 + j * 16;
    float bb = bias[nc];
    #pragma unroll
    for (int r = 0; r < 8; ++r) {
      float cv = acc[j][r] + bb;
      if (RELU) cv = fmaxf(cv, 0.f);
      C[(mr + r) * (long)N + nc] = (OutT)cv;
    }
  }
}

// ---------------- fused 2-way attention ----------------
// One block per (b,sen,seg); wave w = head w (32 lanes x 4 elems = 128 = D_K).
__global__ __launch_bounds__(256)
void k_attention(const float* __restrict__ q,   // [B,SEN,1024]
                 const float* __restrict__ kk,  // [B,SEG,2,1024]
                 const float* __restrict__ vv,  // [B,SEG,2,1024]
                 __bf16* __restrict__ out) {    // [B,SEN,SEG,1024]
  long idx = blockIdx.x;
  int  seg = (int)(idx % NSEG);
  long t   = idx / NSEG;
  int  sen = (int)(t % NSEN);
  int  b   = (int)(t / NSEN);

  const float* qr = q  + ((long)b * NSEN + sen) * DM;
  const float* k0 = kk + (((long)b * NSEG + seg) * 2 + 0) * DM;
  const float* k1 = k0 + DM;
  const float* v0 = vv + (((long)b * NSEG + seg) * 2 + 0) * DM;
  const float* v1 = v0 + DM;

  int d0 = threadIdx.x * 4;
  float4 qv  = *reinterpret_cast<const float4*>(&qr[d0]);
  float4 k0v = *reinterpret_cast<const float4*>(&k0[d0]);
  float4 k1v = *reinterpret_cast<const float4*>(&k1[d0]);
  float p0 = qv.x*k0v.x + qv.y*k0v.y + qv.z*k0v.z + qv.w*k0v.w;
  float p1 = qv.x*k1v.x + qv.y*k1v.y + qv.z*k1v.z + qv.w*k1v.w;
  for (int o = 16; o > 0; o >>= 1) {
    p0 += __shfl_xor(p0, o, 32);
    p1 += __shfl_xor(p1, o, 32);
  }
  const float scale = 0.08838834764831845f;      // 1/sqrt(128)
  float a0 = p0 * scale, a1 = p1 * scale;
  float m  = fmaxf(a0, a1);
  float e0 = __expf(a0 - m), e1 = __expf(a1 - m);
  float inv = 1.f / (e0 + e1);
  float w0 = e0 * inv, w1 = e1 * inv;

  float4 v0v = *reinterpret_cast<const float4*>(&v0[d0]);
  float4 v1v = *reinterpret_cast<const float4*>(&v1[d0]);
  __bf16* o = out + idx * (long)DM + d0;
  o[0] = (__bf16)(w0 * v0v.x + w1 * v1v.x);
  o[1] = (__bf16)(w0 * v0v.y + w1 * v1v.y);
  o[2] = (__bf16)(w0 * v0v.z + w1 * v1v.z);
  o[3] = (__bf16)(w0 * v0v.w + w1 * v1v.w);
}

// ---------------- launch ----------------

extern "C" void kernel_launch(void* const* d_in, const int* in_sizes, int n_in,
                              void* d_out, int out_size, void* d_ws, size_t ws_size,
                              hipStream_t stream) {
  const float* A     = (const float*)d_in[0];
  const float* V     = (const float*)d_in[1];
  const float* S     = (const float*)d_in[2];
  const float* Wq    = (const float*)d_in[3];
  const float* bq    = (const float*)d_in[4];
  const float* Wk    = (const float*)d_in[5];
  const float* bk    = (const float*)d_in[6];
  const float* Wv    = (const float*)d_in[7];
  const float* bv    = (const float*)d_in[8];
  const float* gamma = (const float*)d_in[9];
  const float* beta  = (const float*)d_in[10];
  const float* W1    = (const float*)d_in[11];
  const float* b1    = (const float*)d_in[12];
  const float* W2    = (const float*)d_in[13];
  const float* b2    = (const float*)d_in[14];
  float* out = (float*)d_out;

  char* ws = (char*)d_ws;
  size_t off = 0;
  auto take = [&](size_t bytes) -> char* {
    char* p = ws + off;
    off += (bytes + 255) & ~(size_t)255;
    return p;
  };

  const long AV_ROWS  = (long)BSZ * NSEG;          // 1024
  const long Q_ROWS   = (long)BSZ * NSEN;          // 512
  const long OUT_ROWS = (long)BSZ * NSEN * NSEG;   // 32768

  __bf16* Wq_t = (__bf16*)take((size_t)DM  * DM  * 2);
  __bf16* Wk_t = (__bf16*)take((size_t)DM  * DM  * 2);
  __bf16* Wv_t = (__bf16*)take((size_t)DM  * DM  * 2);
  __bf16* W1_t = (__bf16*)take((size_t)DFF * DM  * 2);   // N=DFF rows x K=DM
  __bf16* W2_t = (__bf16*)take((size_t)DM  * DFF * 2);   // N=DM  rows x K=DFF
  __bf16* avb  = (__bf16*)take((size_t)AV_ROWS * 2 * DM * 2);
  __bf16* Snb  = (__bf16*)take((size_t)Q_ROWS * DM * 2);
  float*  qf   = (float*) take((size_t)Q_ROWS * DM * 4);
  float*  kf   = (float*) take((size_t)AV_ROWS * 2 * DM * 4);
  float*  vf   = (float*) take((size_t)AV_ROWS * 2 * DM * 4);
  __bf16* outb = (__bf16*)take((size_t)OUT_ROWS * DM * 2);
  __bf16* hidb = (__bf16*)take((size_t)OUT_ROWS * DFF * 2);

  // --- prep: weight transposes (fp32 -> bf16, N x K), av interleave, layernorm ---
  {
    long n = (long)DM * DM;
    int g = (int)((n + 255) / 256);
    k_transpose_bf16<<<g, 256, 0, stream>>>(Wq, Wq_t, DM, DM);
    k_transpose_bf16<<<g, 256, 0, stream>>>(Wk, Wk_t, DM, DM);
    k_transpose_bf16<<<g, 256, 0, stream>>>(Wv, Wv_t, DM, DM);
  }
  {
    long n = (long)DM * DFF;
    int g = (int)((n + 255) / 256);
    k_transpose_bf16<<<g, 256, 0, stream>>>(W1, W1_t, DM, DFF);   // -> DFF x DM
    k_transpose_bf16<<<g, 256, 0, stream>>>(W2, W2_t, DFF, DM);   // -> DM x DFF
  }
  {
    long n = AV_ROWS * (long)DM;
    k_build_av<<<(int)((n + 255) / 256), 256, 0, stream>>>(A, V, avb, AV_ROWS, DM);
  }
  k_layernorm_bf16<<<(int)Q_ROWS, 256, 0, stream>>>(S, gamma, beta, Snb, DM);

  // --- projections (fp32 out, small) ---
  k_gemm_bf16nt<0, float><<<dim3(DM / 64, (int)(Q_ROWS / 64)), 128, 0, stream>>>(
      Snb, Wq_t, bq, qf, (int)Q_ROWS, DM, DM);
  k_gemm_bf16nt<0, float><<<dim3(DM / 64, (int)(AV_ROWS * 2 / 64)), 128, 0, stream>>>(
      avb, Wk_t, bk, kf, (int)(AV_ROWS * 2), DM, DM);
  k_gemm_bf16nt<0, float><<<dim3(DM / 64, (int)(AV_ROWS * 2 / 64)), 128, 0, stream>>>(
      avb, Wv_t, bv, vf, (int)(AV_ROWS * 2), DM, DM);

  // --- fused 2-way softmax attention -> bf16 activations ---
  k_attention<<<(int)OUT_ROWS, 256, 0, stream>>>(qf, kf, vf, outb);

  // --- FFN: fc1 (+bias, ReLU, bf16) then fc2 (+bias, fp32 -> d_out) ---
  k_gemm_bf16nt<1, __bf16><<<dim3(DFF / 64, (int)(OUT_ROWS / 64)), 128, 0, stream>>>(
      outb, W1_t, b1, hidb, (int)OUT_ROWS, DFF, DM);
  k_gemm_bf16nt<0, float><<<dim3(DM / 64, (int)(OUT_ROWS / 64)), 128, 0, stream>>>(
      hidb, W2_t, b2, out, (int)OUT_ROWS, DM, DFF);
}
